// ArcMarginLoss_42984032698723
// MI455X (gfx1250) — matrix operand order repaired and checked by reading, not compiled, not run on hardware
//
#include <hip/hip_runtime.h>
#include <hip/hip_bf16.h>
#include <math.h>

typedef __attribute__((ext_vector_type(16))) __bf16 v16bf;
typedef __attribute__((ext_vector_type(8)))  __bf16 v8bf;
typedef __attribute__((ext_vector_type(4)))  __bf16 v4bf;
typedef __attribute__((ext_vector_type(8)))  float  v8f;
typedef __attribute__((ext_vector_type(4)))  int    v4i;

#define B_ROWS 2048
#define DIM    512
#define VOCAB  32000
#define NT     (VOCAB / 128)           // 250 column slabs of 128
#define S_SC   30.0f
#define COS_M  0.8775825618903728f     // cos(0.5)
#define SIN_M  0.4794255386042030f     // sin(0.5)

// ---- gfx1250 async global->LDS copy (guarded; falls back to a plain copy) ----
#if defined(__has_builtin)
#if __has_builtin(__builtin_amdgcn_global_load_async_to_lds_b128)
#define HAVE_ASYNC 1
#endif
#endif
#ifndef HAVE_ASYNC
#define HAVE_ASYNC 0
#endif

typedef __attribute__((address_space(1))) v4i gv4i;
typedef __attribute__((address_space(3))) v4i lv4i;

__device__ __forceinline__ void cp16(const __bf16* g, __bf16* l) {
#if HAVE_ASYNC
  __builtin_amdgcn_global_load_async_to_lds_b128((gv4i*)(v4i*)g, (lv4i*)(v4i*)l, 0, 0);
#else
  *(v8bf*)l = *(const v8bf*)g;
#endif
}

__device__ __forceinline__ void wait_async_all() {
#if HAVE_ASYNC
#if __has_builtin(__builtin_amdgcn_s_wait_asynccnt)
  __builtin_amdgcn_s_wait_asynccnt(0);
#else
  asm volatile("s_wait_asynccnt 0x0" ::: "memory");
#endif
#endif
}

// ---- fused 1/||row|| + normalized bf16 conversion (one wave per row) ---------
__global__ void norm_convert_kernel(const float* __restrict__ mat,
                                    float* __restrict__ rn,
                                    __bf16* __restrict__ outb, int rows) {
  const int lane = threadIdx.x & 31;
  const int row  = blockIdx.x * (blockDim.x >> 5) + (threadIdx.x >> 5);
  if (row >= rows) return;
  const float* p = mat + (size_t)row * DIM;
  float4 v[4];
  float ss = 0.f;
#pragma unroll
  for (int j = 0; j < 4; ++j) {
    v[j] = *(const float4*)(p + lane * 4 + j * 128);
    ss += v[j].x * v[j].x + v[j].y * v[j].y + v[j].z * v[j].z + v[j].w * v[j].w;
  }
#pragma unroll
  for (int off = 16; off > 0; off >>= 1) ss += __shfl_xor(ss, off, 32);
  float r = 1.0f / fmaxf(sqrtf(ss), 1e-12f);
  if (lane == 0) rn[row] = r;
  __bf16* o = outb + (size_t)row * DIM;
#pragma unroll
  for (int j = 0; j < 4; ++j) {
    v4bf b;
    b[0] = (__bf16)(v[j].x * r); b[1] = (__bf16)(v[j].y * r);
    b[2] = (__bf16)(v[j].z * r); b[3] = (__bf16)(v[j].w * r);
    *(v4bf*)(o + lane * 4 + j * 128) = b;
  }
}

// ---- exact f32 cosine at the label column (one wave per row) -----------------
__global__ void cos_y_kernel(const float* __restrict__ x, const float* __restrict__ W,
                             const int* __restrict__ labels,
                             const float* __restrict__ rnx, const float* __restrict__ rnw,
                             float* __restrict__ cosy) {
  const int lane = threadIdx.x & 31;
  const int row  = blockIdx.x * (blockDim.x >> 5) + (threadIdx.x >> 5);
  if (row >= B_ROWS) return;
  const int y = labels[row];
  const float* xp = x + (size_t)row * DIM;
  const float* wp = W + (size_t)y * DIM;
  float d = 0.f;
  for (int k = lane * 4; k < DIM; k += 128) {
    float4 a = *(const float4*)(xp + k);
    float4 b = *(const float4*)(wp + k);
    d += a.x * b.x + a.y * b.y + a.z * b.z + a.w * b.w;
  }
#pragma unroll
  for (int off = 16; off > 0; off >>= 1) d += __shfl_xor(d, off, 32);
  if (lane == 0) cosy[row] = d * rnx[row] * rnw[y];
}

// ---- fused bf16 WMMA GEMM + online softmax partials --------------------------
// Block = 256 threads (8 waves). Tile = 128 rows x 128 cols, K stepped by 32,
// double-buffered LDS staged with async global->LDS copies.
__global__ __launch_bounds__(256)
void arc_gemm_partial_kernel(const __bf16* __restrict__ xb, const __bf16* __restrict__ Wb,
                             float2* __restrict__ partials) {
  __shared__ alignas(32) __bf16 At[2][128 * 32];
  __shared__ alignas(32) __bf16 Bt[2][128 * 32];

  const int tid   = threadIdx.x;
  const int lane  = tid & 31;
  const int warp  = tid >> 5;
  const int mbase = blockIdx.y * 128;
  const int nbase = blockIdx.x * 128;

  // stage one 128x32 bf16 tile of A and B into LDS buffer `buf`
  auto stage = [&](int kk, int buf) {
#pragma unroll
    for (int i = 0; i < 2; ++i) {
      int s   = tid + i * 256;        // 512 16-byte slots per matrix
      int row = s >> 2;               // 4 slots per 32-elem row
      int ko  = (s & 3) << 3;         // 0,8,16,24
      cp16(xb + (size_t)(mbase + row) * DIM + kk + ko, &At[buf][row * 32 + ko]);
      cp16(Wb + (size_t)(nbase + row) * DIM + kk + ko, &Bt[buf][row * 32 + ko]);
    }
  };

  v8f acc[8];
#pragma unroll
  for (int t = 0; t < 8; ++t)
#pragma unroll
    for (int j = 0; j < 8; ++j) acc[t][j] = 0.f;

  const int khalf = (lane >> 4) << 3;          // A: lanes<16 K=0-7,16-23; >=16: 8-15,24-31
  const int arow  = warp * 16 + (lane & 15);
  const int bb    = (lane >> 4) << 3;          // B: lanes<16 K=0-7,16-23; >=16: 8-15,24-31

  stage(0, 0);
  for (int ks = 0; ks < DIM / 32; ++ks) {
    const int buf = ks & 1;
    wait_async_all();
    __syncthreads();
    if (ks + 1 < DIM / 32) stage((ks + 1) * 32, buf ^ 1);

    v16bf af;
    {
      const __bf16* ap = &At[buf][arow * 32];
      v8bf a0 = *(const v8bf*)(ap + khalf);
      v8bf a1 = *(const v8bf*)(ap + khalf + 16);
#pragma unroll
      for (int j = 0; j < 8; ++j) { af[j] = a0[j]; af[j + 8] = a1[j]; }
    }
#pragma unroll
    for (int t = 0; t < 8; ++t) {
      int col  = t * 16 + (lane & 15);
      v16bf bf;
      {
        v8bf b0 = *(const v8bf*)(&Bt[buf][col * 32 + bb]);
        v8bf b1 = *(const v8bf*)(&Bt[buf][col * 32 + bb + 16]);
#pragma unroll
        for (int j = 0; j < 8; ++j) { bf[j] = b0[j]; bf[j + 8] = b1[j]; }
      }
      acc[t] = __builtin_amdgcn_wmma_f32_16x16x32_bf16(false, af, false, bf,
                                                       (short)0, acc[t], false, false);
    }
  }

  // Per-row (max, sumexp) over this 128-col slab of logits s*cos.
  // C layout: VGPR r -> M=r (lanes 0-15) / M=r+8 (lanes 16-31), N = lane&15 + 16*t.
#pragma unroll
  for (int r = 0; r < 8; ++r) {
    float lt[8];
#pragma unroll
    for (int t = 0; t < 8; ++t) lt[t] = S_SC * acc[t][r];
    float m = lt[0];
#pragma unroll
    for (int t = 1; t < 8; ++t) m = fmaxf(m, lt[t]);
#pragma unroll
    for (int off = 1; off < 16; off <<= 1) m = fmaxf(m, __shfl_xor(m, off, 32));
    float se = 0.f;
#pragma unroll
    for (int t = 0; t < 8; ++t) se += __expf(lt[t] - m);
#pragma unroll
    for (int off = 1; off < 16; off <<= 1) se += __shfl_xor(se, off, 32);
    if ((lane & 15) == 0) {
      int rowg = mbase + warp * 16 + r + (lane >> 4) * 8;
      partials[(size_t)rowg * NT + blockIdx.x] = make_float2(m, se);
    }
  }
}

// ---- combine partials + margin correction + mean NLL -------------------------
__global__ void finalize_kernel(const float2* __restrict__ partials,
                                const float* __restrict__ cosy,
                                float* __restrict__ out) {
  __shared__ float red[256];
  const int tid = threadIdx.x;
  float lsum = 0.f;
  for (int i = tid; i < B_ROWS; i += 256) {
    float M = -3.4e38f, S = 0.f;
    const float2* p = partials + (size_t)i * NT;
    for (int j = 0; j < NT; ++j) {
      float2 q = p[j];
      if (q.x > M) { S = S * __expf(M - q.x) + q.y; M = q.x; }
      else         { S += q.y * __expf(q.x - M); }
    }
    float cy  = cosy[i];
    float sy  = sqrtf(fminf(fmaxf(1.f - cy * cy, 0.f), 1.f));
    float phi = cy * COS_M - sy * SIN_M;
    float Smod = S - __expf(S_SC * cy - M) + __expf(S_SC * phi - M);
    Smod = fmaxf(Smod, 1e-30f);
    lsum += logf(Smod) + M - S_SC * phi;
  }
  red[tid] = lsum;
  __syncthreads();
  for (int s = 128; s > 0; s >>= 1) {
    if (tid < s) red[tid] += red[tid + s];
    __syncthreads();
  }
  if (tid == 0) out[0] = red[0] / (float)B_ROWS;
}

extern "C" void kernel_launch(void* const* d_in, const int* in_sizes, int n_in,
                              void* d_out, int out_size, void* d_ws, size_t ws_size,
                              hipStream_t stream) {
  (void)in_sizes; (void)n_in; (void)out_size; (void)ws_size;
  const float* x      = (const float*)d_in[0];
  const float* W      = (const float*)d_in[1];
  const int*   labels = (const int*)d_in[2];
  float* out = (float*)d_out;

  char* ws = (char*)d_ws;
  float*  rnx      = (float*)(ws);                         // 2048
  float*  rnw      = (float*)(ws + 8192);                  // 32000
  float*  cosy     = (float*)(ws + 136192);                // 2048
  float2* partials = (float2*)(ws + 144384);               // 2048*250 float2 = 4 MB
  __bf16* xb       = (__bf16*)(ws + 4240384);              // 2048*512  bf16 = 2 MB
  __bf16* Wb       = (__bf16*)(ws + 6337536);              // 32000*512 bf16 = 32.8 MB

  norm_convert_kernel<<<B_ROWS / 8, 256, 0, stream>>>(x, rnx, xb, B_ROWS);
  norm_convert_kernel<<<VOCAB / 8, 256, 0, stream>>>(W, rnw, Wb, VOCAB);
  cos_y_kernel<<<B_ROWS / 8, 256, 0, stream>>>(x, W, labels, rnx, rnw, cosy);

  dim3 grid(NT, B_ROWS / 128);
  arc_gemm_partial_kernel<<<grid, 256, 0, stream>>>(xb, Wb, partials);

  finalize_kernel<<<1, 256, 0, stream>>>(partials, cosy, out);
}
